// TextClassificationModel_82334523064784
// MI455X (gfx1250) — matrix operand order repaired and checked by reading, not compiled, not run on hardware
//
#include <hip/hip_runtime.h>

// ---------------------------------------------------------------------------
// B=65536, C=64, L=64, D=128.  dots = x @ E^T (M=B, N=C*L, K=D), then
// per-(b,c) max/argmax over L.  Pre-normalize rows into bf16 so the GEMM
// yields cosine directly; bf16 WMMA GEMM with TDM (tensor_load_to_lds)
// staging of LDS tiles (TDM applies the 16B/row pad in hardware);
// workgroup-local max/argmax + LDS count aggregation; float atomics with
// integer-valued addends (deterministic).
// ---------------------------------------------------------------------------

typedef __attribute__((ext_vector_type(16))) __bf16 v16bf;
typedef __attribute__((ext_vector_type(8)))  float  v8f;
typedef __attribute__((ext_vector_type(4)))  unsigned int v4u;
typedef __attribute__((ext_vector_type(8)))  int  v8i;
typedef __attribute__((ext_vector_type(4)))  int  v4i;

union FragBF {
  v16bf v;
  uint4 q[2];
};

#define TILE_M 128
#define TILE_N 64
#define KDIM   128
#define LDA    136   // 128 + 8 bf16 pad (row stride 272 B -> conflict-free)
#define LDC    68    // f32 row stride 272 B

__device__ __forceinline__ unsigned short f32_to_bf16_rne(float f) {
  unsigned int u = __float_as_uint(f);
  u += 0x7FFFu + ((u >> 16) & 1u);
  return (unsigned short)(u >> 16);
}

// TDM: DMA a tile_rows x 128 bf16 tile (row-major, row stride 128) from
// global into LDS at lds_off, inserting 16B padding after each 256B row
// (pad_interval=5 -> 64 DWORDs, pad_amount=3 -> 4 DWORDs) => LDS stride 272B.
__device__ __forceinline__ void tdm_load_tile_bf16(
    unsigned lds_off, const void* gptr, int tile_rows, int nrows_total) {
  unsigned long long ga = (unsigned long long)(uintptr_t)gptr;
  // D# group 0: count=1 | lds_addr | global_addr[56:0] | type=2
  v4u g0 = { 1u,
             lds_off,
             (unsigned)ga,
             ((unsigned)((ga >> 32) & 0x01FFFFFFu)) | (2u << 30) };
  // D# group 1: data_size=1(2B), pad_enable, pad_interval=5, pad_amount=3;
  // tensor_dim0=128, tensor_dim1=nrows_total, tile_dim0=128,
  // tile_dim1=tile_rows, tensor_dim0_stride=128.
  v8i g1 = { (int)((1u << 16) | (1u << 20) | (5u << 22) | (3u << 25)),
             (int)(128u << 16),
             (int)(((unsigned)nrows_total & 0xFFFFu) << 16),
             (int)((((unsigned)nrows_total >> 16) & 0xFFFFu) | (128u << 16)),
             (int)((unsigned)tile_rows & 0xFFFFu),
             128, 0, 0 };
  v4i gz = { 0, 0, 0, 0 };
#if defined(__clang_major__) && __clang_major__ >= 23
  v8i gz8 = { 0, 0, 0, 0, 0, 0, 0, 0 };
  __builtin_amdgcn_tensor_load_to_lds(g0, g1, gz, gz, gz8, 0);
#else
  __builtin_amdgcn_tensor_load_to_lds(g0, g1, gz, gz, 0);
#endif
}

// out_bf16[row,:] = in[row,:] / max(||in[row,:]||, 1e-8), rows of length 128.
__global__ __launch_bounds__(256) void rownorm_bf16_kernel(
    const float* __restrict__ in, unsigned short* __restrict__ out, int nrows) {
  const int row  = blockIdx.x * 8 + (threadIdx.x >> 5);
  const int lane = threadIdx.x & 31;
  if (row >= nrows) return;
  const float4 v = *(const float4*)(in + (size_t)row * 128 + lane * 4);
  float s = v.x * v.x + v.y * v.y + v.z * v.z + v.w * v.w;
#pragma unroll
  for (int m = 16; m > 0; m >>= 1) s += __shfl_xor(s, m, 32);
  const float inv = 1.0f / fmaxf(sqrtf(s), 1e-8f);
  ushort4 o;
  o.x = f32_to_bf16_rne(v.x * inv);
  o.y = f32_to_bf16_rne(v.y * inv);
  o.z = f32_to_bf16_rne(v.z * inv);
  o.w = f32_to_bf16_rne(v.w * inv);
  *(ushort4*)(out + (size_t)row * 128 + lane * 4) = o;
}

__global__ void init_counts_kernel(const float* __restrict__ leaf_in,
                                   float* __restrict__ counts, int n) {
  int i = blockIdx.x * blockDim.x + threadIdx.x;
  if (i < n) counts[i] = leaf_in[i];
}

// Grid: (B/128, C). Block: 256 threads = 8 waves, 4(M) x 2(N) wave grid.
// Each wave: 2x2 tiles of v_wmma_f32_16x16x32_bf16, K-loop of 4.
__global__ __launch_bounds__(256) void cosmax_gemm_kernel(
    const unsigned short* __restrict__ xb,   // [B,128] bf16 (pre-normalized)
    const unsigned short* __restrict__ eb,   // [C*L,128] bf16 (pre-normalized)
    float* __restrict__ cosMax,              // [B,C]
    float* __restrict__ counts) {            // [C,L]
  __shared__ __align__(16) unsigned char smem[(TILE_M + TILE_N) * LDA * 2];
  unsigned short* ldsA = (unsigned short*)smem;              // 128 x LDA bf16
  unsigned short* ldsB = ldsA + TILE_M * LDA;                //  64 x LDA bf16
  float*          ldsC = (float*)smem;                       // 128 x LDC f32 (reused)
  unsigned int*   cnt  = (unsigned int*)(smem + TILE_M * LDC * 4);

  const int b0   = blockIdx.x * TILE_M;
  const int c    = blockIdx.y;
  const int t    = threadIdx.x;
  const int lane = t & 31;
  const int wave = t >> 5;
  const int wm   = wave >> 1;          // 0..3 -> M offset wm*32
  const int wn   = wave & 1;           // 0..1 -> N offset wn*32
  const int half = lane >> 4;          // lane half selects K sub-block
  const int lr   = lane & 15;

  // ---- TDM-stage both tiles (one wave issues; EXEC/lanes irrelevant to TDM)
  if (t == 0) {
    const unsigned ldsA_off = (unsigned)(uintptr_t)(void*)ldsA;
    const unsigned ldsB_off = (unsigned)(uintptr_t)(void*)ldsB;
    tdm_load_tile_bf16(ldsA_off, xb + (size_t)b0 * KDIM, TILE_M,
                       (int)(gridDim.x * TILE_M));
    tdm_load_tile_bf16(ldsB_off, eb + (size_t)c * TILE_N * KDIM, TILE_N,
                       (int)(gridDim.y * TILE_N));
  }
  __builtin_amdgcn_s_wait_tensorcnt(0);   // no-op for waves with TENSORcnt==0
  __syncthreads();

  v8f zero = {};
  v8f acc[2][2];
  acc[0][0] = zero; acc[0][1] = zero; acc[1][0] = zero; acc[1][1] = zero;

#pragma unroll
  for (int ks = 0; ks < 4; ++ks) {
    const int k0 = ks * 32;
    FragBF a[2], b[2];
    // A 16x32 bf16: lanes<16: M=lr, K=k0..k0+15;
    //               lanes>=16: same M, K=k0+16..k0+31 (contiguous, matches B).
#pragma unroll
    for (int i = 0; i < 2; ++i) {
      const unsigned short* p = ldsA + (wm * 32 + i * 16 + lr) * LDA + k0 + half * 16;
      a[i].q[0] = *(const uint4*)p;
      a[i].q[1] = *(const uint4*)(p + 8);
    }
    // B 32x16 bf16: lane = N (mod 16); lanes 0-15 K=k0..k0+15,
    //               lanes 16-31 K=k0+16..k0+31 (contiguous 32B per lane).
#pragma unroll
    for (int j = 0; j < 2; ++j) {
      const unsigned short* p = ldsB + (wn * 32 + j * 16 + lr) * LDA + k0 + half * 16;
      b[j].q[0] = *(const uint4*)p;
      b[j].q[1] = *(const uint4*)(p + 8);
    }
#pragma unroll
    for (int i = 0; i < 2; ++i)
#pragma unroll
      for (int j = 0; j < 2; ++j)
        acc[i][j] = __builtin_amdgcn_wmma_f32_16x16x32_bf16(
            false, a[i].v, false, b[j].v, (short)0, acc[i][j], false, false);
  }
  __syncthreads();  // done reading ldsA/ldsB; reuse region as ldsC

  // D layout (16x16 f32): VGPR v -> M = v + (lane<16 ? 0 : 8), N = lane%16.
#pragma unroll
  for (int i = 0; i < 2; ++i)
#pragma unroll
    for (int j = 0; j < 2; ++j)
#pragma unroll
      for (int v = 0; v < 8; ++v) {
        const int row = wm * 32 + i * 16 + v + half * 8;
        const int col = wn * 32 + j * 16 + lr;
        ldsC[row * LDC + col] = acc[i][j][v];
      }
  if (t >= 128 && t < 192) cnt[t - 128] = 0;
  __syncthreads();

  // Per-row max + first-argmax over 64 leaves (matches jnp.argmax ties).
  if (t < TILE_M) {
    const float* rowp = ldsC + t * LDC;
    float best = -3.0e38f;
    int bi = 0;
#pragma unroll
    for (int cc = 0; cc < TILE_N; cc += 4) {
      float4 q = *(const float4*)(rowp + cc);
      if (q.x > best) { best = q.x; bi = cc; }
      if (q.y > best) { best = q.y; bi = cc + 1; }
      if (q.z > best) { best = q.z; bi = cc + 2; }
      if (q.w > best) { best = q.w; bi = cc + 3; }
    }
    cosMax[(size_t)(b0 + t) * gridDim.y + c] = best;
    atomicAdd(&cnt[bi], 1u);   // LDS aggregation
  }
  __syncthreads();

  if (t < TILE_N) {
    const unsigned int n = cnt[t];
    if (n) atomicAdd(&counts[c * TILE_N + t], (float)n);  // exact integer adds
  }
}

extern "C" void kernel_launch(void* const* d_in, const int* in_sizes, int n_in,
                              void* d_out, int out_size, void* d_ws, size_t ws_size,
                              hipStream_t stream) {
  const float* x    = (const float*)d_in[0];  // [B,128]
  const float* emb  = (const float*)d_in[1];  // [C,L,128]
  const float* leaf = (const float*)d_in[2];  // [C,L]

  const int D  = 128;
  const int B  = in_sizes[0] / D;       // 65536
  const int CL = in_sizes[2];           // 4096
  const int C  = 64;

  unsigned short* xb = (unsigned short*)d_ws;          // B*128 bf16
  unsigned short* eb = xb + (size_t)B * D;             // CL*128 bf16

  float* cosMax = (float*)d_out;                       // [B,C]
  float* counts = cosMax + (size_t)B * C;              // [C,L]

  init_counts_kernel<<<(CL + 255) / 256, 256, 0, stream>>>(leaf, counts, CL);
  rownorm_bf16_kernel<<<(B + 7) / 8, 256, 0, stream>>>(x, xb, B);
  rownorm_bf16_kernel<<<(CL + 7) / 8, 256, 0, stream>>>(emb, eb, CL);

  dim3 grid(B / TILE_M, C);
  cosmax_gemm_kernel<<<grid, 256, 0, stream>>>(xb, eb, cosMax, counts);
}
